// AdaConv_12025908428895
// MI455X (gfx1250) — matrix operand order repaired and checked by reading, not compiled, not run on hardware
//
#include <hip/hip_runtime.h>
#include <hip/hip_bf16.h>

typedef __attribute__((ext_vector_type(16))) _Float16 v16h;
typedef __attribute__((ext_vector_type(8)))  float    v8f;
typedef __attribute__((ext_vector_type(4)))  float    v4f;

#define N_SAMP 8
#define C_IN   256
#define HW     128
#define SD     512
#define PP     4          // channels per group
#define KDIM   2048       // 512 * 2 * 2
#define MROWS  1024       // c_out * p filters
#define NCOLS  72         // 8 samples * 9 positions
#define NCOLS_PAD 80
#define PER_SAMPLE (C_IN*HW*HW)   // 4194304

// ---------------- stats: per-sample mean / var (deterministic 2-stage) ---------
__global__ __launch_bounds__(256) void stats_partial_kernel(
    const float* __restrict__ pred, float* __restrict__ partials) {
  const int n = blockIdx.y;
  const size_t base = (size_t)n * PER_SAMPLE + (size_t)blockIdx.x * 4096;
  float s = 0.f, sq = 0.f;
#pragma unroll
  for (int i = 0; i < 4; ++i) {
    const float4 v = *(const float4*)(pred + base + (size_t)i * 1024 + threadIdx.x * 4);
    s  += v.x + v.y + v.z + v.w;
    sq += v.x * v.x + v.y * v.y + v.z * v.z + v.w * v.w;
  }
  __shared__ float ss[256], ssq[256];
  ss[threadIdx.x] = s; ssq[threadIdx.x] = sq;
  __syncthreads();
  for (int o = 128; o > 0; o >>= 1) {
    if (threadIdx.x < o) {
      ss[threadIdx.x]  += ss[threadIdx.x + o];
      ssq[threadIdx.x] += ssq[threadIdx.x + o];
    }
    __syncthreads();
  }
  if (threadIdx.x == 0) {
    partials[(size_t)(n * 1024 + blockIdx.x) * 2 + 0] = ss[0];
    partials[(size_t)(n * 1024 + blockIdx.x) * 2 + 1] = ssq[0];
  }
}

__global__ __launch_bounds__(256) void stats_finalize_kernel(
    const float* __restrict__ partials, float* __restrict__ stats) {
  const int n = blockIdx.x;
  float s = 0.f, sq = 0.f;
  for (int i = threadIdx.x; i < 1024; i += 256) {
    s  += partials[(size_t)(n * 1024 + i) * 2 + 0];
    sq += partials[(size_t)(n * 1024 + i) * 2 + 1];
  }
  __shared__ float ss[256], ssq[256];
  ss[threadIdx.x] = s; ssq[threadIdx.x] = sq;
  __syncthreads();
  for (int o = 128; o > 0; o >>= 1) {
    if (threadIdx.x < o) {
      ss[threadIdx.x]  += ss[threadIdx.x + o];
      ssq[threadIdx.x] += ssq[threadIdx.x + o];
    }
    __syncthreads();
  }
  if (threadIdx.x == 0) { stats[2 * n] = ss[0]; stats[2 * n + 1] = ssq[0]; }
}

// ---------------- pooled = mean over 4x4 style ---------------------------------
__global__ __launch_bounds__(256) void pooled_kernel(
    const float* __restrict__ style, float* __restrict__ pooled) {
  const int t = blockIdx.x * 256 + threadIdx.x;       // 8*512 = 4096 exact
  const float* p = style + (size_t)t * 16;
  float s = 0.f;
#pragma unroll
  for (int i = 0; i < 16; ++i) s += p[i];
  pooled[t] = s * (1.f / 16.f);
}

// ---------------- pointwise kernel & bias generation (tiny matmuls) ------------
__global__ __launch_bounds__(256) void pw_kernel(
    const float* __restrict__ pooled,
    const float* __restrict__ pk_w, const float* __restrict__ pk_b,
    const float* __restrict__ pb_w, const float* __restrict__ pb_b,
    float* __restrict__ pwkn, float* __restrict__ pwbias) {
  const int t = blockIdx.x * 256 + threadIdx.x;       // 8*1280 = 10240 exact
  const int n = t / 1280, j = t % 1280;
  const float* pr = pooled + n * SD;
  if (j < 1024) {
    const float* w = pk_w + (size_t)j * SD;
    float a = pk_b[j];
#pragma unroll 4
    for (int k = 0; k < SD; ++k) a += pr[k] * w[k];
    pwkn[n * 1024 + j] = a;
  } else {
    const int oc = j - 1024;
    const float* w = pb_w + (size_t)oc * SD;
    float a = pb_b[oc];
#pragma unroll 4
    for (int k = 0; k < SD; ++k) a += pr[k] * w[k];
    pwbias[n * C_IN + oc] = a;
  }
}

// ---------------- pack B (style patches) into WMMA B-fragment layout -----------
// B[k][col], k=(c*4+ky*2+kx), col=n*9+(py*3+px). Fragment: lane holds col=lane&15,
// K = kt*32 + (lane>>4)*16 + e  (contiguous k order, same as the A fragment).
__global__ __launch_bounds__(256) void bfrag_kernel(
    const float* __restrict__ style, _Float16* __restrict__ bfrag) {
  const int t = blockIdx.x * 256 + threadIdx.x;       // 5*64*32*16 = 163840 exact
  const int e    = t & 15;
  const int lane = (t >> 4) & 31;
  const int kt   = (t >> 9) & 63;
  const int ct   = t >> 15;
  const int k    = kt * 32 + ((lane >> 4) * 16) + e;
  const int col  = ct * 16 + (lane & 15);
  float v = 0.f;
  if (col < NCOLS) {
    const int n  = col / 9, pos = col % 9;
    const int py = pos / 3, px = pos % 3;
    const int c  = k >> 2, ky = (k >> 1) & 1, kx = k & 1;
    v = style[(((size_t)n * SD + c) * 4 + (py + ky)) * 4 + (px + kx)];
  }
  bfrag[t] = (_Float16)v;
}

// ---------------- WMMA GEMM: depthwise kernel generation -----------------------
// C[1024 x 72] = dw_w[1024 x 2048] * stylepatch[2048 x 72], f16 in / f32 acc.
__global__ __launch_bounds__(32) void stylegemm_wmma(
    const float* __restrict__ dw_w, const _Float16* __restrict__ bfrag,
    float* __restrict__ gemmC) {
  const int lane = threadIdx.x & 31;
  const int mt = blockIdx.x;                // 0..63
  const int ct = blockIdx.y;                // 0..4
  const int half = lane >> 4;
  const int mrow = mt * 16 + (lane & 15);
  const float* Arow = dw_w + (size_t)mrow * KDIM + half * 16;
  const _Float16* Bp = bfrag + ((size_t)(ct * 64) * 32 + lane) * 16;
  v8f acc = {};
  for (int kt = 0; kt < 64; ++kt) {
    // A 16-bit 16x32 fragment: lane m=lane&15; K runs kbase+half*16+[0,16).
    const v4f x0 = *(const v4f*)(Arow + kt * 32);
    const v4f x1 = *(const v4f*)(Arow + kt * 32 + 4);
    const v4f x2 = *(const v4f*)(Arow + kt * 32 + 8);
    const v4f x3 = *(const v4f*)(Arow + kt * 32 + 12);
    v16h a;
#pragma unroll
    for (int e = 0; e < 4; ++e) {
      a[e]      = (_Float16)x0[e];
      a[e + 4]  = (_Float16)x1[e];
      a[e + 8]  = (_Float16)x2[e];
      a[e + 12] = (_Float16)x3[e];
    }
    const v16h b = *(const v16h*)(Bp + (size_t)kt * 32 * 16);
    acc = __builtin_amdgcn_wmma_f32_16x16x32_f16(
        false, a, false, b, (short)0, acc, false, false);
  }
  const int col = ct * 16 + (lane & 15);
  if (col < NCOLS) {
#pragma unroll
    for (int j = 0; j < 8; ++j) {
      const int row = mt * 16 + half * 8 + j;   // D: vgpr j -> M=j (+8 for hi half)
      gemmC[(size_t)row * NCOLS_PAD + col] = acc[j];
    }
  }
}

// ---------------- fold pointwise into depthwise: Keff --------------------------
__global__ __launch_bounds__(256) void keff_kernel(
    const float* __restrict__ gemmC, const float* __restrict__ dw_b,
    const float* __restrict__ pwkn, float* __restrict__ keff) {
  const int t = blockIdx.x * 256 + threadIdx.x;       // 8*256*36 = 73728 exact
  const int n = t / 9216;
  const int r = t % 9216;
  const int oc = r / 36;
  const int r2 = r % 36;
  const int i2 = r2 / 9, tap = r2 % 9;
  const int base = (oc >> 2) << 2;
  float a = 0.f;
#pragma unroll
  for (int i = 0; i < 4; ++i) {
    const int ch = (base + i) * 4 + i2;
    a += pwkn[n * 1024 + oc * 4 + i] *
         (gemmC[(size_t)ch * NCOLS_PAD + n * 9 + tap] + dw_b[ch]);
  }
  keff[(size_t)(n * 64 + (oc >> 2)) * 144 + (oc & 3) * 36 + i2 * 9 + tap] = a;
}

// ---------------- main fused conv: norm + reflect pad + 3x3 grouped + bias -----
__global__ __launch_bounds__(256) void conv_main(
    const float* __restrict__ pred, const float* __restrict__ keff,
    const float* __restrict__ pwbias, const float* __restrict__ stats,
    const int* __restrict__ normflag, float* __restrict__ out) {
  const int g = blockIdx.z;                 // n*64 + c4
  const int n = g >> 6, c4 = g & 63;
  __shared__ float tile[4][34][34];

  float mu = 0.f, istd = 1.f;
  if (normflag[0]) {
    const float m  = stats[2 * n]     * (1.f / (float)PER_SAMPLE);
    const float sq = stats[2 * n + 1] * (1.f / (float)PER_SAMPLE);
    mu = m;
    istd = rsqrtf((sq - m * m) + 1e-5f);
  }

  // Keff + bias: uniform addresses -> scalar loads / SGPR residency.
  float kr[144];
#pragma unroll
  for (int i = 0; i < 144; ++i) kr[i] = keff[(size_t)g * 144 + i];
  const float b0 = pwbias[n * C_IN + c4 * 4 + 0];
  const float b1 = pwbias[n * C_IN + c4 * 4 + 1];
  const float b2 = pwbias[n * C_IN + c4 * 4 + 2];
  const float b3 = pwbias[n * C_IN + c4 * 4 + 3];

  const int ty0 = blockIdx.y * 32 - 1, tx0 = blockIdx.x * 32 - 1;
  const size_t chbase = ((size_t)n * C_IN + c4 * 4) * (HW * HW);

  // Stage 4 channels x 34x34 with reflect padding + normalization applied.
  for (int i = threadIdx.x; i < 4 * 34 * 34; i += 256) {
    const int ch = i / 1156;
    const int rr = i - ch * 1156;
    const int ly = rr / 34, lx = rr - ly * 34;
    int gy = ty0 + ly; gy = gy < 0 ? -gy : (gy > 127 ? 254 - gy : gy);
    int gx = tx0 + lx; gx = gx < 0 ? -gx : (gx > 127 ? 254 - gx : gx);
    const float v = pred[chbase + (size_t)ch * (HW * HW) + gy * HW + gx];
    tile[ch][ly][lx] = (v - mu) * istd;
  }
  __syncthreads();

#pragma unroll
  for (int q = 0; q < 4; ++q) {
    const int pi = threadIdx.x + q * 256;   // 0..1023 -> 32x32 tile
    const int py = pi >> 5, px = pi & 31;
    float a0 = b0, a1 = b1, a2 = b2, a3 = b3;
#pragma unroll
    for (int i2 = 0; i2 < 4; ++i2) {
#pragma unroll
      for (int dy = 0; dy < 3; ++dy) {
#pragma unroll
        for (int dx = 0; dx < 3; ++dx) {
          const float xv = tile[i2][py + dy][px + dx];
          const int t = i2 * 9 + dy * 3 + dx;
          a0 += xv * kr[t];
          a1 += xv * kr[36 + t];
          a2 += xv * kr[72 + t];
          a3 += xv * kr[108 + t];
        }
      }
    }
    const size_t po = (size_t)(ty0 + 1 + py) * HW + (tx0 + 1 + px);
    out[chbase + (size_t)0 * (HW * HW) + po] = a0;
    out[chbase + (size_t)1 * (HW * HW) + po] = a1;
    out[chbase + (size_t)2 * (HW * HW) + po] = a2;
    out[chbase + (size_t)3 * (HW * HW) + po] = a3;
  }
}

// ---------------- launcher -----------------------------------------------------
extern "C" void kernel_launch(void* const* d_in, const int* in_sizes, int n_in,
                              void* d_out, int out_size, void* d_ws, size_t ws_size,
                              hipStream_t stream) {
  (void)in_sizes; (void)n_in; (void)out_size; (void)ws_size;
  const float* style = (const float*)d_in[0];
  const float* pred  = (const float*)d_in[1];
  const float* dw_w  = (const float*)d_in[2];
  const float* dw_b  = (const float*)d_in[3];
  const float* pk_w  = (const float*)d_in[4];
  const float* pk_b  = (const float*)d_in[5];
  const float* pb_w  = (const float*)d_in[6];
  const float* pb_b  = (const float*)d_in[7];
  const int*   nrm   = (const int*)d_in[8];
  float* out = (float*)d_out;
  float* ws  = (float*)d_ws;

  // Workspace layout (float offsets; bfrag 32B-aligned for v16h loads)
  float* stats    = ws + 0;           // 16
  float* partials = ws + 1024;        // 16384
  float* pooled   = ws + 18432;       // 4096
  float* pwkn     = ws + 22528;       // 8192
  float* pwbias   = ws + 30720;       // 2048
  float* gemmC    = ws + 32768;       // 81920 (1024 x 80)
  float* keff     = ws + 114688;      // 73728
  _Float16* bfrag = (_Float16*)(ws + 188416);  // 163840 halfs

  stats_partial_kernel<<<dim3(1024, 8), 256, 0, stream>>>(pred, partials);
  stats_finalize_kernel<<<8, 256, 0, stream>>>(partials, stats);
  pooled_kernel<<<16, 256, 0, stream>>>(style, pooled);
  pw_kernel<<<40, 256, 0, stream>>>(pooled, pk_w, pk_b, pb_w, pb_b, pwkn, pwbias);
  bfrag_kernel<<<640, 256, 0, stream>>>(style, bfrag);
  stylegemm_wmma<<<dim3(64, 5), 32, 0, stream>>>(dw_w, bfrag, gemmC);
  keff_kernel<<<288, 256, 0, stream>>>(gemmC, dw_b, pwkn, keff);
  conv_main<<<dim3(4, 4, 512), 256, 0, stream>>>(pred, keff, pwbias, stats, nrm, out);
}
